// Architecture_17205638987791
// MI455X (gfx1250) — matrix operand order repaired and checked by reading, not compiled, not run on hardware
//
#include <hip/hip_runtime.h>
#include <hip/hip_bf16.h>
#include <math.h>
#include <stdint.h>

typedef __bf16 bf16;
typedef __attribute__((ext_vector_type(16))) __bf16 bf16x16;
typedef __attribute__((ext_vector_type(8)))  __bf16 bf16x8;
typedef __attribute__((ext_vector_type(8)))  float  f32x8;
typedef unsigned int u32x4 __attribute__((ext_vector_type(4)));
typedef int          i32x4 __attribute__((ext_vector_type(4)));
typedef int          i32x8 __attribute__((ext_vector_type(8)));

#define D_MODEL 512
#define SEQ_L   1024
#define BATCH   4
#define NEIG    24
#define NLAYER  4
#define ROWS    (BATCH * SEQ_L)      /* 4096 */
#define KD_PHI  (NEIG * D_MODEL)     /* 12288 */
#define NFFT    2048
#define NFREQ   1025

#if defined(__AMDGCN__) && __has_builtin(__builtin_amdgcn_tensor_load_to_lds)
#define STU_HAS_TDM 1
#else
#define STU_HAS_TDM 0
#endif

/* ------------------------------------------------------------------ */
/* TDM: 2-D tile DMA global -> LDS with row padding (pitch 64+8 bf16)  */
/* D# group0: count=1 | lds_addr | global_addr | type=2                */
/* D# group1: data_size=1(2B), pad_enable, pad_interval=4 (32 DWORDs   */
/*            per row), pad_amount=3 (4 DWORDs = 8 bf16)               */
/* ------------------------------------------------------------------ */
__device__ __forceinline__ unsigned stu_sgpr(unsigned v)
{
    return (unsigned)__builtin_amdgcn_readfirstlane((int)v);
}

__device__ __forceinline__ void stu_tdm_load_tile_2d(
    unsigned lds_off, const void* gptr,
    unsigned dim0, unsigned dim1,          /* tensor extent (elements)  */
    unsigned tile0, unsigned tile1,        /* tile extent (elements)    */
    unsigned stride0)                      /* row stride (elements)     */
{
#if STU_HAS_TDM
    unsigned long long ga = (unsigned long long)(uintptr_t)gptr;
    u32x4 g0;
    g0[0] = stu_sgpr(1u);                                   /* count=1 */
    g0[1] = stu_sgpr(lds_off);
    g0[2] = stu_sgpr((unsigned)ga);
    g0[3] = stu_sgpr(((unsigned)(ga >> 32) & 0x01FFFFFFu) | (2u << 30));
    i32x8 g1;
    g1[0] = (int)stu_sgpr((1u << 16) | (1u << 20) | (4u << 22) | (3u << 25));
    g1[1] = (int)stu_sgpr((dim0 & 0xFFFFu) << 16);          /* dim0 @ bit48 */
    g1[2] = (int)stu_sgpr(((dim0 >> 16) & 0xFFFFu) | ((dim1 & 0xFFFFu) << 16));
    g1[3] = (int)stu_sgpr(((dim1 >> 16) & 0xFFFFu) | ((tile0 & 0xFFFFu) << 16));
    g1[4] = (int)stu_sgpr(tile1 & 0xFFFFu);                 /* tile_dim2 = 0 */
    g1[5] = (int)stu_sgpr(stride0);                         /* stride0[31:0] */
    g1[6] = 0;
    g1[7] = 0;
    i32x4 z4 = {0, 0, 0, 0};
#if __clang_major__ >= 23
    i32x8 z8 = {0, 0, 0, 0, 0, 0, 0, 0};
    __builtin_amdgcn_tensor_load_to_lds(g0, g1, z4, z4, z8, 0);
#else
    __builtin_amdgcn_tensor_load_to_lds(g0, g1, z4, z4, 0);
#endif
#else
    (void)lds_off; (void)gptr; (void)dim0; (void)dim1;
    (void)tile0; (void)tile1; (void)stride0;
#endif
}

/* ------------------------------------------------------------------ */
/* WMMA bf16 GEMM:  C[M,N] = A[M,Kd] (bf16,row-major) @ Bt[N,Kd]^T (+bias)
   Tile 128x128, K-step 64, 8 waves, each wave: 2(M) x 4(N) wmma tiles.
   A/B tiles staged in LDS by the Tensor Data Mover (wave 0 issues the
   descriptors, waits TENSORcnt, barrier releases the WMMA waves).      */
/* ------------------------------------------------------------------ */
#define TM 128
#define TN 128
#define TK 64
#define LDP 72   /* LDS pitch in bf16 elements (64 data + 8 pad) */

__global__ void __launch_bounds__(256)
stu_gemm_bf16(const bf16* __restrict__ A, const bf16* __restrict__ Bt,
              float* __restrict__ C, const float* __restrict__ bias,
              int M, int N, int Kd)
{
    __shared__ alignas(16) bf16 As[TM * LDP];
    __shared__ alignas(16) bf16 Bs[TN * LDP];

    const int tid  = threadIdx.x;
    const int wave = tid >> 5;
    const int lane = tid & 31;
    const int bm   = blockIdx.y * TM;
    const int bn   = blockIdx.x * TN;
    const int wm   = (wave >> 1) * 32;   /* 0,32,64,96 */
    const int wn   = (wave & 1) * 64;    /* 0,64       */

    f32x8 acc[2][4];
    const f32x8 zero = {0.f,0.f,0.f,0.f,0.f,0.f,0.f,0.f};
    for (int i = 0; i < 2; ++i)
        for (int j = 0; j < 4; ++j)
            acc[i][j] = zero;

#if STU_HAS_TDM
    const unsigned as_base = (unsigned)(uintptr_t)&As[0];
    const unsigned bs_base = (unsigned)(uintptr_t)&Bs[0];
    const int wv = __builtin_amdgcn_readfirstlane((int)threadIdx.x) >> 5;
#endif

    for (int k0 = 0; k0 < Kd; k0 += TK) {
#if STU_HAS_TDM
        /* Tensor Data Mover: DMA both tiles straight into LDS (padded rows) */
        if (wv == 0) {
            stu_tdm_load_tile_2d(as_base, &A[(size_t)bm * Kd + k0],
                                 (unsigned)Kd, (unsigned)M, TK, TM, (unsigned)Kd);
            stu_tdm_load_tile_2d(bs_base, &Bt[(size_t)bn * Kd + k0],
                                 (unsigned)Kd, (unsigned)N, TK, TN, (unsigned)Kd);
            __builtin_amdgcn_s_wait_tensorcnt(0);
        }
        __syncthreads();
#else
        /* fallback: manual staging, 128 rows x 64 cols each, b128 chunks */
        #pragma unroll
        for (int c = 0; c < 4; ++c) {
            int chunk = tid + c * 256;          /* 0..1023 */
            int r   = chunk >> 3;
            int col = (chunk & 7) * 8;
            *(bf16x8*)&As[r * LDP + col] =
                *(const bf16x8*)&A[(size_t)(bm + r) * Kd + k0 + col];
            *(bf16x8*)&Bs[r * LDP + col] =
                *(const bf16x8*)&Bt[(size_t)(bn + r) * Kd + k0 + col];
        }
        __syncthreads();
#endif

        #pragma unroll
        for (int kk = 0; kk < TK; kk += 32) {
            union Frag { bf16x16 v; bf16x8 h[2]; };

            /* A fragments: lane 0-15 = row M, elems 0-7 = K+0..7, 8-15 = K+16..23;
               lanes 16-31 get K+8..15 / K+24..31                                  */
            const int arow  = lane & 15;
            const int ahalf = lane >> 4;
            Frag af[2];
            #pragma unroll
            for (int tm = 0; tm < 2; ++tm) {
                const bf16* p = &As[(wm + tm * 16 + arow) * LDP + kk + ahalf * 8];
                af[tm].h[0] = *(const bf16x8*)p;
                af[tm].h[1] = *(const bf16x8*)(p + 16);
            }

            /* B fragments: lane 0-15 = col N, elems 0-7 = K+0..7, 8-15 = K+16..23;
               lanes 16-31 get K+8..15 / K+24..31                                  */
            Frag bfr[4];
            #pragma unroll
            for (int tn = 0; tn < 4; ++tn) {
                const bf16* p = &Bs[(wn + tn * 16 + (lane & 15)) * LDP + kk + (lane >> 4) * 8];
                bfr[tn].h[0] = *(const bf16x8*)p;
                bfr[tn].h[1] = *(const bf16x8*)(p + 16);
            }

            #pragma unroll
            for (int tm = 0; tm < 2; ++tm)
                #pragma unroll
                for (int tn = 0; tn < 4; ++tn)
                    acc[tm][tn] = __builtin_amdgcn_wmma_f32_16x16x32_bf16(
                        false, af[tm].v, false, bfr[tn].v,
                        (short)0, acc[tm][tn], false, false);
        }
        __syncthreads();
    }

    /* epilogue: C/D layout col = lane&15, row = (lane>>4)*8 + r */
    for (int tm = 0; tm < 2; ++tm) {
        const int row0 = bm + wm + tm * 16 + (lane >> 4) * 8;
        for (int tn = 0; tn < 4; ++tn) {
            const int col = bn + wn + tn * 16 + (lane & 15);
            const float bv = bias ? bias[col] : 0.f;
            #pragma unroll
            for (int r = 0; r < 8; ++r)
                C[(size_t)(row0 + r) * N + col] = acc[tm][tn][r] + bv;
        }
    }
}

/* ------------------------------------------------------------------ */
/* conversion / packing kernels                                        */
/* ------------------------------------------------------------------ */
__global__ void __launch_bounds__(256)
stu_f32_to_bf16(const float* __restrict__ src, bf16* __restrict__ dst, int n)
{
    int i = blockIdx.x * 256 + threadIdx.x;
    if (i < n) dst[i] = (bf16)src[i];
}

/* src [R][C] f32  ->  dst [C][R] bf16 (transpose) */
__global__ void __launch_bounds__(256)
stu_transpose_bf16(const float* __restrict__ src, bf16* __restrict__ dst, int R, int C)
{
    int i = blockIdx.x * 256 + threadIdx.x;
    if (i < R * C) {
        int r = i / C, c = i % C;
        dst[(size_t)c * R + r] = (bf16)src[i];
    }
}

/* m_u[l]: [o][d][ku] f32 -> dst[(ku*512+o)][d] bf16 */
__global__ void __launch_bounds__(256)
stu_mu_pack(const float* __restrict__ src, bf16* __restrict__ dst)
{
    int i = blockIdx.x * 256 + threadIdx.x;          /* < 3*512*512 */
    int ku  = i / (512 * 512);
    int rem = i % (512 * 512);
    int o = rem / 512, d = rem % 512;
    dst[(size_t)(ku * 512 + o) * 512 + d] = (bf16)src[(size_t)(o * 512 + d) * 3 + ku];
}

/* ------------------------------------------------------------------ */
/* LayerNorm over D=512; emits fp32 + bf16                             */
/* ------------------------------------------------------------------ */
__global__ void __launch_bounds__(256)
stu_layernorm(const float* __restrict__ x, const float* __restrict__ sc,
              const float* __restrict__ bi, float* __restrict__ h,
              bf16* __restrict__ hbf)
{
    __shared__ float red[256];
    const int row = blockIdx.x, tid = threadIdx.x;
    const float* xr = x + (size_t)row * 512;
    float v0 = xr[tid], v1 = xr[tid + 256];
    red[tid] = v0 + v1;
    __syncthreads();
    for (int s = 128; s > 0; s >>= 1) { if (tid < s) red[tid] += red[tid + s]; __syncthreads(); }
    float mu = red[0] * (1.f / 512.f);
    __syncthreads();
    float d0 = v0 - mu, d1 = v1 - mu;
    red[tid] = d0 * d0 + d1 * d1;
    __syncthreads();
    for (int s = 128; s > 0; s >>= 1) { if (tid < s) red[tid] += red[tid + s]; __syncthreads(); }
    float rs = rsqrtf(red[0] * (1.f / 512.f) + 1e-5f);
    float o0 = d0 * rs * sc[tid] + bi[tid];
    float o1 = d1 * rs * sc[tid + 256] + bi[tid + 256];
    h[(size_t)row * 512 + tid]        = o0;
    h[(size_t)row * 512 + tid + 256]  = o1;
    hbf[(size_t)row * 512 + tid]       = (bf16)o0;
    hbf[(size_t)row * 512 + tid + 256] = (bf16)o1;
}

/* ------------------------------------------------------------------ */
/* Stockham radix-2 FFT, n=2048, in LDS (ping-pong), 512 threads       */
/* ------------------------------------------------------------------ */
__device__ __forceinline__ void stu_fft2048(float* R0, float* I0, float* R1, float* I1,
                                            int tid, float sign,
                                            float** outR, float** outI)
{
    float *sR = R0, *sI = I0, *dR = R1, *dI = I1;
    for (int Ns = 1; Ns < NFFT; Ns <<= 1) {
        for (int j = tid; j < NFFT / 2; j += 512) {
            int r = j & (Ns - 1);
            float aR = sR[j],            aI = sI[j];
            float bR = sR[j + NFFT / 2], bI = sI[j + NFFT / 2];
            float ang = sign * 3.14159265358979323846f * (float)r / (float)Ns;
            float sn, cs;
            __sincosf(ang, &sn, &cs);
            float tR = cs * bR - sn * bI;
            float tI = cs * bI + sn * bR;
            int dd = 2 * j - r;                 /* (j/Ns)*2Ns + r */
            dR[dd]      = aR + tR;  dI[dd]      = aI + tI;
            dR[dd + Ns] = aR - tR;  dI[dd + Ns] = aI - tI;
        }
        __syncthreads();
        float* t;
        t = sR; sR = dR; dR = t;
        t = sI; sI = dI; dI = t;
    }
    *outR = sR; *outI = sI;
}

/* forward rfft of h[b,:,d] (zero-padded to 2048) -> U[b,f,d] (f<=1024) */
__global__ void __launch_bounds__(512)
stu_fft_fwd(const float* __restrict__ h, float2* __restrict__ U)
{
    __shared__ float R0[NFFT], I0[NFFT], R1[NFFT], I1[NFFT];
    const int b = blockIdx.x / 512, d = blockIdx.x % 512, tid = threadIdx.x;
    for (int i = tid; i < NFFT; i += 512) {
        R0[i] = (i < SEQ_L) ? h[((size_t)b * SEQ_L + i) * 512 + d] : 0.f;
        I0[i] = 0.f;
    }
    __syncthreads();
    float *sR, *sI;
    stu_fft2048(R0, I0, R1, I1, tid, -1.f, &sR, &sI);
    for (int f = tid; f < NFREQ; f += 512) {
        float2 v; v.x = sR[f]; v.y = sI[f];
        U[((size_t)b * NFREQ + f) * 512 + d] = v;
    }
}

/* rfft of eig filters * eigval^0.25 -> Vp[k][f] */
__global__ void __launch_bounds__(512)
stu_fft_filters(const float* __restrict__ ev, const float* __restrict__ evals,
                float2* __restrict__ Vp)
{
    __shared__ float R0[NFFT], I0[NFFT], R1[NFFT], I1[NFFT];
    const int k = blockIdx.x, tid = threadIdx.x;
    for (int i = tid; i < NFFT; i += 512) {
        R0[i] = (i < SEQ_L) ? ev[(size_t)i * NEIG + k] : 0.f;
        I0[i] = 0.f;
    }
    __syncthreads();
    float *sR, *sI;
    stu_fft2048(R0, I0, R1, I1, tid, -1.f, &sR, &sI);
    float scale = powf(evals[k], 0.25f);
    for (int f = tid; f < NFREQ; f += 512) {
        float2 v; v.x = sR[f] * scale; v.y = sI[f] * scale;
        Vp[(size_t)k * NFREQ + f] = v;
    }
}

/* inverse: S = U[b,:,d] * Vp[k,:] (Hermitian-extended), ifft, first L reals
   -> xt bf16 at [(b*L+t)][k*512+d]                                          */
__global__ void __launch_bounds__(512)
stu_fft_inv(const float2* __restrict__ U, const float2* __restrict__ Vp,
            bf16* __restrict__ xt)
{
    __shared__ float R0[NFFT], I0[NFFT], R1[NFFT], I1[NFFT];
    const int d = blockIdx.x % 512;
    const int k = (blockIdx.x / 512) % NEIG;
    const int b = blockIdx.x / (512 * NEIG);
    const int tid = threadIdx.x;
    for (int i = tid; i < NFFT; i += 512) {
        int f = (i <= 1024) ? i : (NFFT - i);
        float2 u = U[((size_t)b * NFREQ + f) * 512 + d];
        float2 v = Vp[(size_t)k * NFREQ + f];
        float re = u.x * v.x - u.y * v.y;
        float im = u.x * v.y + u.y * v.x;
        if (i > 1024) im = -im;              /* conjugate mirror */
        R0[i] = re; I0[i] = im;
    }
    __syncthreads();
    float *sR, *sI;
    stu_fft2048(R0, I0, R1, I1, tid, +1.f, &sR, &sI);
    const float inv_n = 1.f / (float)NFFT;
    for (int t = tid; t < SEQ_L; t += 512)
        xt[(size_t)(b * SEQ_L + t) * KD_PHI + k * 512 + d] = (bf16)(sR[t] * inv_n);
}

/* ------------------------------------------------------------------ */
/* deltas += sum_i shift_i(tmp[:, i*512 + o])                          */
/* ------------------------------------------------------------------ */
__global__ void __launch_bounds__(256)
stu_combine_ar(float* __restrict__ deltas, const float* __restrict__ tmp)
{
    int i = blockIdx.x * 256 + threadIdx.x;          /* < ROWS*512 */
    int row = i >> 9, o = i & 511;
    int l = row & (SEQ_L - 1);
    float v = deltas[i];
    v += tmp[(size_t)row * 1536 + o];
    if (l >= 1) v += tmp[(size_t)(row - 1) * 1536 + 512 + o];
    if (l >= 2) v += tmp[(size_t)(row - 2) * 1536 + 1024 + o];
    deltas[i] = v;
}

/* ------------------------------------------------------------------ */
/* sequential recurrence y_t = M0 y_{t-1} + M1 y_{t-2} + delta_t       */
/* one workgroup per batch; state in LDS; m_y row o = [M0 row, M1 row] */
/* ------------------------------------------------------------------ */
__global__ void __launch_bounds__(512)
stu_recurrence(const float* __restrict__ deltas, const float* __restrict__ my,
               float* __restrict__ ys)
{
    __shared__ float y1[512], y2[512];
    const int b = blockIdx.x, o = threadIdx.x;
    y1[o] = 0.f; y2[o] = 0.f;
    __syncthreads();
    const float4* Wr = (const float4*)&my[(size_t)o * 1024];  /* 256 float4 */
    for (int t = 0; t < SEQ_L; ++t) {
        float acc = deltas[((size_t)b * SEQ_L + t) * 512 + o];
        if (t + 8 < SEQ_L)
            __builtin_prefetch(&deltas[((size_t)b * SEQ_L + t + 8) * 512 + o], 0, 1);
        #pragma unroll 8
        for (int j = 0; j < 128; ++j) {
            float4 w = Wr[j];
            float4 yv = *(const float4*)&y1[j * 4];
            acc += w.x * yv.x + w.y * yv.y + w.z * yv.z + w.w * yv.w;
        }
        #pragma unroll 8
        for (int j = 0; j < 128; ++j) {
            float4 w = Wr[128 + j];
            float4 yv = *(const float4*)&y2[j * 4];
            acc += w.x * yv.x + w.y * yv.y + w.z * yv.z + w.w * yv.w;
        }
        __syncthreads();
        y2[o] = y1[o];
        y1[o] = acc;
        ys[((size_t)b * SEQ_L + t) * 512 + o] = acc;
        __syncthreads();
    }
}

/* ------------------------------------------------------------------ */
/* GELU(erf) -> bf16 ;  GLU + residual -> fp32 + bf16                  */
/* ------------------------------------------------------------------ */
__global__ void __launch_bounds__(256)
stu_gelu(const float* __restrict__ ys, bf16* __restrict__ gbf)
{
    int i = blockIdx.x * 256 + threadIdx.x;
    float v = ys[i];
    float g = 0.5f * v * (1.f + erff(v * 0.70710678118654752f));
    gbf[i] = (bf16)g;
}

__global__ void __launch_bounds__(256)
stu_glu_residual(const float* __restrict__ g, float* __restrict__ x,
                 bf16* __restrict__ xbf)
{
    int i = blockIdx.x * 256 + threadIdx.x;          /* < ROWS*512 */
    int row = i >> 9, o = i & 511;
    float a  = g[(size_t)row * 1024 + o];
    float bb = g[(size_t)row * 1024 + 512 + o];
    float hh = a / (1.f + expf(-bb));
    float xn = hh + x[i];
    x[i]   = xn;
    xbf[i] = (bf16)xn;
}

/* ------------------------------------------------------------------ */
/* host launcher                                                       */
/* ------------------------------------------------------------------ */
extern "C" void kernel_launch(void* const* d_in, const int* in_sizes, int n_in,
                              void* d_out, int out_size, void* d_ws, size_t ws_size,
                              hipStream_t stream)
{
    (void)in_sizes; (void)n_in; (void)out_size; (void)ws_size;

    const float* in_x     = (const float*)d_in[0];
    const float* emb_w    = (const float*)d_in[1];
    const float* emb_b    = (const float*)d_in[2];
    const float* ln_scale = (const float*)d_in[3];
    const float* ln_bias  = (const float*)d_in[4];
    const float* m_y      = (const float*)d_in[5];
    const float* m_u      = (const float*)d_in[6];
    const float* m_phi    = (const float*)d_in[7];
    const float* w1       = (const float*)d_in[8];
    const float* b1       = (const float*)d_in[9];
    const float* proj_w   = (const float*)d_in[10];
    const float* proj_b   = (const float*)d_in[11];
    const float* eig_vals = (const float*)d_in[12];
    const float* eig_vecs = (const float*)d_in[13];

    char* base = (char*)d_ws;
    size_t off = 0;
    auto wsAlloc = [&](size_t bytes) -> char* {
        char* p = base + off;
        off += (bytes + 255) & ~(size_t)255;
        return p;
    };
    bf16*   embT   = (bf16*)  wsAlloc((size_t)512 * 512 * 2);
    bf16*   projT  = (bf16*)  wsAlloc((size_t)512 * 512 * 2);
    bf16*   mphiT  = (bf16*)  wsAlloc((size_t)512 * KD_PHI * 2);
    bf16*   muT    = (bf16*)  wsAlloc((size_t)1536 * 512 * 2);
    bf16*   w1T    = (bf16*)  wsAlloc((size_t)1024 * 512 * 2);
    bf16*   xbf    = (bf16*)  wsAlloc((size_t)ROWS * 512 * 2);
    float*  x      = (float*) wsAlloc((size_t)ROWS * 512 * 4);
    float*  h      = (float*) wsAlloc((size_t)ROWS * 512 * 4);
    bf16*   hbf    = (bf16*)  wsAlloc((size_t)ROWS * 512 * 2);
    float2* U      = (float2*)wsAlloc((size_t)BATCH * NFREQ * 512 * 8);
    float2* Vp     = (float2*)wsAlloc((size_t)NEIG * NFREQ * 8);
    bf16*   xt     = (bf16*)  wsAlloc((size_t)ROWS * KD_PHI * 2);
    float*  deltas = (float*) wsAlloc((size_t)ROWS * 512 * 4);
    float*  tmp    = (float*) wsAlloc((size_t)ROWS * 1536 * 4);
    float*  ys     = (float*) wsAlloc((size_t)ROWS * 512 * 4);
    bf16*   gbf    = (bf16*)  wsAlloc((size_t)ROWS * 512 * 2);
    float*  g      = (float*) wsAlloc((size_t)ROWS * 1024 * 4);

    const int flat512 = (ROWS * 512) / 256;

    /* static weight prep */
    stu_transpose_bf16<<<(512 * 512) / 256, 256, 0, stream>>>(emb_w,  embT,  512, 512);
    stu_transpose_bf16<<<(512 * 512) / 256, 256, 0, stream>>>(proj_w, projT, 512, 512);
    stu_f32_to_bf16<<<flat512, 256, 0, stream>>>(in_x, xbf, ROWS * 512);

    /* embedding GEMM */
    {
        dim3 grid(512 / TN, ROWS / TM);
        stu_gemm_bf16<<<grid, 256, 0, stream>>>(xbf, embT, x, emb_b, ROWS, 512, 512);
    }

    /* spectral filters (once) */
    stu_fft_filters<<<NEIG, 512, 0, stream>>>(eig_vecs, eig_vals, Vp);

    for (int l = 0; l < NLAYER; ++l) {
        /* per-layer weight prep */
        stu_transpose_bf16<<<(KD_PHI * 512) / 256, 256, 0, stream>>>(
            m_phi + (size_t)l * KD_PHI * 512, mphiT, KD_PHI, 512);
        stu_mu_pack<<<(3 * 512 * 512) / 256, 256, 0, stream>>>(
            m_u + (size_t)l * 512 * 512 * 3, muT);
        stu_transpose_bf16<<<(512 * 1024) / 256, 256, 0, stream>>>(
            w1 + (size_t)l * 512 * 1024, w1T, 512, 1024);

        /* layernorm */
        stu_layernorm<<<ROWS, 256, 0, stream>>>(x, ln_scale + l * 512, ln_bias + l * 512, h, hbf);

        /* spectral causal conv (FFT) */
        stu_fft_fwd<<<BATCH * 512, 512, 0, stream>>>(h, U);
        stu_fft_inv<<<BATCH * NEIG * 512, 512, 0, stream>>>(U, Vp, xt);

        /* delta_phi = xt @ m_phi */
        {
            dim3 grid(512 / TN, ROWS / TM);
            stu_gemm_bf16<<<grid, 256, 0, stream>>>(xt, mphiT, deltas, nullptr, ROWS, 512, KD_PHI);
        }
        /* AR projections (3 weights fused into N=1536) */
        {
            dim3 grid(1536 / TN, ROWS / TM);
            stu_gemm_bf16<<<grid, 256, 0, stream>>>(hbf, muT, tmp, nullptr, ROWS, 1536, 512);
        }
        stu_combine_ar<<<flat512, 256, 0, stream>>>(deltas, tmp);

        /* sequential y recurrence */
        stu_recurrence<<<BATCH, 512, 0, stream>>>(deltas, m_y + (size_t)l * 512 * 1024, ys);

        /* gelu -> w1 GEMM -> glu + residual */
        stu_gelu<<<flat512, 256, 0, stream>>>(ys, gbf);
        {
            dim3 grid(1024 / TN, ROWS / TM);
            stu_gemm_bf16<<<grid, 256, 0, stream>>>(gbf, w1T, g, b1 + l * 1024, ROWS, 1024, 512);
        }
        stu_glu_residual<<<flat512, 256, 0, stream>>>(g, x, xbf);
    }

    /* output projection -> d_out (fp32) */
    {
        dim3 grid(512 / TN, ROWS / TM);
        stu_gemm_bf16<<<grid, 256, 0, stream>>>(xbf, projT, (float*)d_out, proj_b, ROWS, 512, 512);
    }
}
